// CrossSemanticAttentionModule0_6743098655543
// MI455X (gfx1250) — matrix operand order repaired and checked by reading, not compiled, not run on hardware
//
#include <hip/hip_runtime.h>

// CDNA5 / gfx1250 implementation of the cross-modal attention block.
// All GEMMs (conv-as-implicit-GEMM, q/k/v/up projections, QK^T scores,
// V*P^T) run on v_wmma_f32_16x16x32_f16 with f32 accumulation.
// Attention uses a 2-pass flash scheme (row stats, then streamed P chunks
// through LDS) so the 134MB N x N matrix is never materialized.
// Conv zero-padding uses a zeroed guard row + per-lane pointer select
// (v_cndmask on address) instead of divergent control flow, keeping the
// inner loop a straight-line load clause + WMMA.

typedef __attribute__((ext_vector_type(16))) _Float16 v16h;
typedef __attribute__((ext_vector_type(8)))  _Float16 v8h;
typedef __attribute__((ext_vector_type(8)))  float    v8f;

#define BATCH 2
#define CIN   512
#define CD    256
#define CQ    32
#define NPIX  4096
#define IMGW  64
#define BN_EPS 1e-5f
#define THREADS 256

// ---------------------------------------------------------------- helpers

__device__ __forceinline__ v8f wmma_f16(v16h a, v16h b, v8f c) {
  // D = A(16x32) * B(32x16) + C, f32 accum
  return __builtin_amdgcn_wmma_f32_16x16x32_f16(
      false, a, false, b, (short)0, c, false, false);
}

__device__ __forceinline__ v8f zero8f() {
  v8f z;
#pragma unroll
  for (int i = 0; i < 8; ++i) z[i] = 0.0f;
  return z;
}

// A fragment: M=16 rows (lane&15), K=32. Row-major source [row][k], k contig.
// k order used here (contiguous, not ISA table order): lanes 0-15 hold K 0..15; lanes 16-31 hold K 16..31.
__device__ __forceinline__ v16h load_a_frag(const _Float16* base, int stride,
                                            int lane) {
  const int row = lane & 15, sel = lane >> 4;
  const _Float16* p = base + (size_t)row * stride + sel * 16;
  v8h lo = *(const v8h*)p;        // K = 16*sel + 0..7
  v8h hi = *(const v8h*)(p + 8);  // K = 16*sel + 8..15
  v16h r;
#pragma unroll
  for (int i = 0; i < 8; ++i) { r[i] = lo[i]; r[i + 8] = hi[i]; }
  return r;
}

// B fragment: K=32 rows, N=16 cols (lane&15). Source row-major [col][k].
// k order used here (contiguous, matches A): lanes 0-15 hold K 0..15 of their column, lanes 16-31 K 16..31.
__device__ __forceinline__ v16h load_b_frag(const _Float16* base, int stride,
                                            int lane) {
  const int col = lane & 15, sel = lane >> 4;
  const _Float16* p = base + (size_t)col * stride + sel * 16;
  v8h lo = *(const v8h*)p;
  v8h hi = *(const v8h*)(p + 8);
  v16h r;
#pragma unroll
  for (int i = 0; i < 8; ++i) { r[i] = lo[i]; r[i + 8] = hi[i]; }
  return r;
}

// ---------------------------------------------------------------- packing

// f32 NCHW [B][CIN][N] -> f16 pixel-major [B][N][CIN]
__global__ void __launch_bounds__(THREADS)
pack_x_kernel(const float* __restrict__ x, _Float16* __restrict__ xt) {
  size_t i = (size_t)blockIdx.x * blockDim.x + threadIdx.x;
  size_t total = (size_t)BATCH * CIN * NPIX;
  if (i >= total) return;
  int c = (int)(i % CIN);
  size_t t = i / CIN;
  int p = (int)(t % NPIX);
  int b = (int)(t / NPIX);
  xt[i] = (_Float16)x[((size_t)b * CIN + c) * NPIX + p];
}

// f32 [CD][CIN][3][3] -> f16 [9][CD][CIN]
__global__ void __launch_bounds__(THREADS)
pack_wconv_kernel(const float* __restrict__ w, _Float16* __restrict__ wt) {
  size_t i = (size_t)blockIdx.x * blockDim.x + threadIdx.x;
  size_t total = (size_t)9 * CD * CIN;
  if (i >= total) return;
  int c = (int)(i % CIN);
  size_t t = i / CIN;
  int o = (int)(t % CD);
  int tap = (int)(t / CD);
  wt[i] = (_Float16)w[((size_t)o * CIN + c) * 9 + tap];
}

// generic f32 -> f16 copy (row-major weights)
__global__ void __launch_bounds__(THREADS)
pack_mat_kernel(const float* __restrict__ src, _Float16* __restrict__ dst,
                int n) {
  int i = blockIdx.x * blockDim.x + threadIdx.x;
  if (i < n) dst[i] = (_Float16)src[i];
}

// zero the conv guard row (padding source)
__global__ void __launch_bounds__(THREADS)
zero_guard_kernel(_Float16* __restrict__ g, int n) {
  int i = blockIdx.x * blockDim.x + threadIdx.x;
  if (i < n) g[i] = (_Float16)0.0f;
}

// fold conv bias + BN into per-channel scale/shift
__global__ void __launch_bounds__(THREADS)
fuse_bn_kernel(const float* __restrict__ cb, const float* __restrict__ g,
               const float* __restrict__ bb, const float* __restrict__ bm,
               const float* __restrict__ bv, float* __restrict__ A,
               float* __restrict__ Bc) {
  int c = blockIdx.x * blockDim.x + threadIdx.x;
  if (c >= CD) return;
  float s = g[c] * rsqrtf(bv[c] + BN_EPS);
  A[c] = s;
  Bc[c] = cb[c] * s + (bb[c] - bm[c] * s);
}

// ------------------------------------------------- conv3x3 + BN + PReLU

// out[b, co, p] = prelu(bn(sum_t sum_cin W[t][co][cin] * x[b][shift_t(p)][cin]))
// stored pixel-major f16: conv_t[b][p][co]
__global__ void __launch_bounds__(THREADS)
conv_kernel(const _Float16* __restrict__ xt, const _Float16* __restrict__ wt,
            const _Float16* __restrict__ guard,
            const float* __restrict__ A, const float* __restrict__ Bc,
            const float* __restrict__ prelu_a, _Float16* __restrict__ conv_t) {
  int wave = (int)((blockIdx.x * blockDim.x + threadIdx.x) >> 5);
  int lane = threadIdx.x & 31;
  int pt = wave & 255;          // NPIX/16 = 256 pixel tiles
  int rest = wave >> 8;
  int ct = rest & 15;           // CD/16 = 16 channel tiles
  int b = rest >> 4;
  if (b >= BATCH) return;

  const int col = lane & 15, sel = lane >> 4;
  const int p0 = pt * 16;
  const int y = p0 >> 6;        // 16 | 64, so a tile shares one image row
  const int x0 = p0 & 63;

  v8f acc = zero8f();

#pragma unroll
  for (int ky = 0; ky < 3; ++ky) {
#pragma unroll
    for (int kx = 0; kx < 3; ++kx) {
      const int tap = ky * 3 + kx;
      const int ys = y + ky - 1;
      const int xs = x0 + col + kx - 1;
      const bool valid = (ys >= 0) & (ys < IMGW) & (xs >= 0) & (xs < IMGW);
      // per-lane pointer select: OOB lanes read the zeroed guard row.
      const int src_pix = valid ? (ys * IMGW + xs) : 0;
      const _Float16* rowp =
          valid ? (xt + ((size_t)b * NPIX + src_pix) * CIN) : guard;
      const _Float16* abase = wt + ((size_t)tap * CD + ct * 16) * CIN;
      for (int kc = 0; kc < CIN; kc += 32) {
        const _Float16* p = rowp + kc + sel * 16;
        v8h lo = *(const v8h*)p;
        v8h hi = *(const v8h*)(p + 8);
        v16h bf;
#pragma unroll
        for (int i = 0; i < 8; ++i) { bf[i] = lo[i]; bf[i + 8] = hi[i]; }
        v16h af = load_a_frag(abase + kc, CIN, lane);
        acc = wmma_f16(af, bf, acc);
      }
    }
  }

  const float a = prelu_a[0];
  const int pix = p0 + col;
  const int cbase = ct * 16 + sel * 8;
  v8h out;
#pragma unroll
  for (int r = 0; r < 8; ++r) {
    int c = cbase + r;
    float z = acc[r] * A[c] + Bc[c];
    z = z > 0.0f ? z : a * z;
    out[r] = (_Float16)z;
  }
  *(v8h*)(conv_t + ((size_t)b * NPIX + pix) * CD + cbase) = out;
}

// ------------------------------------------------- 1x1 projections (GEMM)

// out_t[b][p][o] = W[o][k] . in_t[b][p][k] + bias[o]   (pixel-major f16 out)
__global__ void __launch_bounds__(THREADS)
gemm_pm_kernel(const _Float16* __restrict__ Wt, const float* __restrict__ bias,
               const _Float16* __restrict__ in_t, _Float16* __restrict__ out_t,
               int O, int K) {
  int wave = (int)((blockIdx.x * blockDim.x + threadIdx.x) >> 5);
  int lane = threadIdx.x & 31;
  int pt = wave & 255;
  int rest = wave >> 8;
  int otiles = O >> 4;
  int ot = rest % otiles;
  int b = rest / otiles;
  if (b >= BATCH) return;

  const int col = lane & 15, sel = lane >> 4;
  v8f acc = zero8f();
  const _Float16* abase = Wt + (size_t)(ot * 16) * K;
  const _Float16* bbase = in_t + ((size_t)b * NPIX + pt * 16) * K;
  for (int kc = 0; kc < K; kc += 32) {
    v16h af = load_a_frag(abase + kc, K, lane);
    v16h bf = load_b_frag(bbase + kc, K, lane);
    acc = wmma_f16(af, bf, acc);
  }
  const int pix = pt * 16 + col;
  const int obase = ot * 16 + sel * 8;
  v8h out;
#pragma unroll
  for (int r = 0; r < 8; ++r) out[r] = (_Float16)(acc[r] + bias[obase + r]);
  *(v8h*)(out_t + ((size_t)b * NPIX + pix) * O + obase) = out;
}

// out_cm[b][o][p] = W[o][k] . in_t[b][p][k] + bias[o]  (channel-major f16 out)
__global__ void __launch_bounds__(THREADS)
gemm_cm_kernel(const _Float16* __restrict__ Wt, const float* __restrict__ bias,
               const _Float16* __restrict__ in_t, _Float16* __restrict__ out_cm,
               int O, int K) {
  int wave = (int)((blockIdx.x * blockDim.x + threadIdx.x) >> 5);
  int lane = threadIdx.x & 31;
  int pt = wave & 255;
  int rest = wave >> 8;
  int otiles = O >> 4;
  int ot = rest % otiles;
  int b = rest / otiles;
  if (b >= BATCH) return;

  const int col = lane & 15, sel = lane >> 4;
  v8f acc = zero8f();
  const _Float16* abase = Wt + (size_t)(ot * 16) * K;
  const _Float16* bbase = in_t + ((size_t)b * NPIX + pt * 16) * K;
  for (int kc = 0; kc < K; kc += 32) {
    v16h af = load_a_frag(abase + kc, K, lane);
    v16h bf = load_b_frag(bbase + kc, K, lane);
    acc = wmma_f16(af, bf, acc);
  }
  const int pix = pt * 16 + col;
  const int obase = ot * 16 + sel * 8;
#pragma unroll
  for (int r = 0; r < 8; ++r) {
    int o = obase + r;
    out_cm[((size_t)b * O + o) * NPIX + pix] = (_Float16)(acc[r] + bias[o]);
  }
}

// ------------------------------------------------- attention pass 1: stats

// per query i: m_i = max_j q.k, l_i = sum_j exp(s - m_i)  (online softmax)
// s^T tile: A = K tile [j x cq], B = Q tile [cq x i]; D lane = query column.
__global__ void __launch_bounds__(THREADS)
stats_kernel(const _Float16* __restrict__ q_t, const _Float16* __restrict__ k_t,
             float* __restrict__ m_out, float* __restrict__ l_out) {
  int wave = (int)((blockIdx.x * blockDim.x + threadIdx.x) >> 5);
  int lane = threadIdx.x & 31;
  int qt = wave & 255;
  int b = wave >> 8;
  if (b >= BATCH) return;

  v16h qf = load_b_frag(q_t + ((size_t)b * NPIX + qt * 16) * CQ, CQ, lane);
  float m = -1e30f, l = 0.0f;
  for (int j = 0; j < NPIX; j += 16) {
    v16h kf = load_a_frag(k_t + ((size_t)b * NPIX + j) * CQ, CQ, lane);
    v8f s = wmma_f16(kf, qf, zero8f());
    float mx = s[0];
#pragma unroll
    for (int r = 1; r < 8; ++r) mx = fmaxf(mx, s[r]);
    float mn = fmaxf(m, mx);
    float add = 0.0f;
#pragma unroll
    for (int r = 0; r < 8; ++r) add += __expf(s[r] - mn);
    l = l * __expf(m - mn) + add;
    m = mn;
  }
  // lanes L and L+16 hold the same query (different j rows) -> combine
  float mp = __shfl_xor(m, 16, 32);
  float lp = __shfl_xor(l, 16, 32);
  float ma = fmaxf(m, mp);
  float la = l * __expf(m - ma) + lp * __expf(mp - ma);
  if (lane < 16) {
    int i = qt * 16 + lane;
    m_out[(size_t)b * NPIX + i] = ma;
    l_out[(size_t)b * NPIX + i] = la;
  }
}

// --------------------------------------- attention pass 2: o = V P^T (flash)

// Block = 8 waves, owns one (b, 16-query) tile and all 16 channel tiles.
// Per 256-key block: each wave computes one 32-key P chunk (2 score WMMAs +
// exp, normalized by stats) into LDS, then all waves stream 8 chunks as
// B-fragments into their 2 channel-tile accumulators (16 o-WMMAs/wave).
__global__ void __launch_bounds__(THREADS)
attn_o_kernel(const _Float16* __restrict__ q_t, const _Float16* __restrict__ k_t,
              const _Float16* __restrict__ v_cm,
              const _Float16* __restrict__ conv_t,
              const float* __restrict__ m_s, const float* __restrict__ l_s,
              const float* __restrict__ gamma_p, _Float16* __restrict__ o_t) {
  __shared__ __align__(16) _Float16 p_lds[16][264];  // [query][256 keys]+pad

  const int lane = threadIdx.x & 31;
  const int wv = threadIdx.x >> 5;  // 0..7
  const int qt = blockIdx.x & 255;
  const int b = blockIdx.x >> 8;
  const int col = lane & 15, sel = lane >> 4;
  const int qi = qt * 16 + col;

  const float mi = m_s[(size_t)b * NPIX + qi];
  const float invl = 1.0f / l_s[(size_t)b * NPIX + qi];
  const float gamma = gamma_p[0];

  v16h qf = load_b_frag(q_t + ((size_t)b * NPIX + qt * 16) * CQ, CQ, lane);
  v8f acc0 = zero8f(), acc1 = zero8f();
  const int ct0 = wv, ct1 = wv + 8;

  for (int jb = 0; jb < NPIX; jb += 256) {
    // -------- phase 1: this wave's 32-key probability chunk -> LDS
    const int j0 = jb + wv * 32;
    v16h kf1 = load_a_frag(k_t + ((size_t)b * NPIX + j0) * CQ, CQ, lane);
    v16h kf2 = load_a_frag(k_t + ((size_t)b * NPIX + j0 + 16) * CQ, CQ, lane);
    v8f s1 = wmma_f16(kf1, qf, zero8f());   // rows j0..j0+15
    v8f s2 = wmma_f16(kf2, qf, zero8f());   // rows j0+16..j0+31
    v8h pa, pb;
#pragma unroll
    for (int r = 0; r < 8; ++r) {
      float s1p = __shfl_xor(s1[r], 16, 32);
      float s2p = __shfl_xor(s2[r], 16, 32);
      // lanes<16 cover local keys 0..15, lanes>=16 cover 16..31
      float sa = (sel == 0) ? s1[r] : s2p;  // local j = 16*sel + r
      float sb = (sel == 0) ? s1p : s2[r];  // local j = 16*sel + 8 + r
      pa[r] = (_Float16)(__expf(sa - mi) * invl);
      pb[r] = (_Float16)(__expf(sb - mi) * invl);
    }
    _Float16* dst = &p_lds[col][wv * 32 + sel * 16];
    *(v8h*)dst = pa;
    *(v8h*)(dst + 8) = pb;
    __syncthreads();

    // -------- phase 2: o accumulation over the 8 chunks of this key block
#pragma unroll
    for (int cc = 0; cc < 8; ++cc) {
      const _Float16* ps = &p_lds[col][cc * 32 + sel * 16];
      v8h plo = *(const v8h*)ps;
      v8h phi = *(const v8h*)(ps + 8);
      v16h pf;
#pragma unroll
      for (int i = 0; i < 8; ++i) { pf[i] = plo[i]; pf[i + 8] = phi[i]; }
      v16h va = load_a_frag(
          v_cm + ((size_t)b * CD + ct0 * 16) * NPIX + jb + cc * 32, NPIX, lane);
      acc0 = wmma_f16(va, pf, acc0);
      v16h vb = load_a_frag(
          v_cm + ((size_t)b * CD + ct1 * 16) * NPIX + jb + cc * 32, NPIX, lane);
      acc1 = wmma_f16(vb, pf, acc1);
    }
    __syncthreads();
  }

  // epilogue: o = gamma * o + conv, store pixel-major f16
#pragma unroll
  for (int half = 0; half < 2; ++half) {
    const int ct = half ? ct1 : ct0;
    const v8f& acc = half ? acc1 : acc0;
    const int cbase = ct * 16 + sel * 8;
    v8h cv = *(const v8h*)(conv_t + ((size_t)b * NPIX + qi) * CD + cbase);
    v8h out;
#pragma unroll
    for (int r = 0; r < 8; ++r)
      out[r] = (_Float16)(gamma * acc[r] + (float)cv[r]);
    *(v8h*)(o_t + ((size_t)b * NPIX + qi) * CD + cbase) = out;
  }
}

// ------------------------------------------------- up-projection + residual

__global__ void __launch_bounds__(THREADS)
up_kernel(const _Float16* __restrict__ Wt, const float* __restrict__ bias,
          const _Float16* __restrict__ o_t, const float* __restrict__ x_in,
          float* __restrict__ out) {
  int wave = (int)((blockIdx.x * blockDim.x + threadIdx.x) >> 5);
  int lane = threadIdx.x & 31;
  int pt = wave & 255;
  int rest = wave >> 8;
  int ot = rest & 31;           // CIN/16 = 32
  int b = rest >> 5;
  if (b >= BATCH) return;

  const int col = lane & 15, sel = lane >> 4;
  v8f acc = zero8f();
  const _Float16* abase = Wt + (size_t)(ot * 16) * CD;
  const _Float16* bbase = o_t + ((size_t)b * NPIX + pt * 16) * CD;
  for (int kc = 0; kc < CD; kc += 32) {
    v16h af = load_a_frag(abase + kc, CD, lane);
    v16h bf = load_b_frag(bbase + kc, CD, lane);
    acc = wmma_f16(af, bf, acc);
  }
  const int pix = pt * 16 + col;
  const int obase = ot * 16 + sel * 8;
#pragma unroll
  for (int r = 0; r < 8; ++r) {
    int co = obase + r;
    size_t idx = ((size_t)b * CIN + co) * NPIX + pix;
    out[idx] = acc[r] + bias[co] + x_in[idx];
  }
}

// ---------------------------------------------------------------- launcher

extern "C" void kernel_launch(void* const* d_in, const int* in_sizes, int n_in,
                              void* d_out, int out_size, void* d_ws,
                              size_t ws_size, hipStream_t stream) {
  (void)in_sizes; (void)n_in; (void)out_size; (void)ws_size;

  const float* input[2] = {(const float*)d_in[0], (const float*)d_in[1]};
  struct ModP {
    const float *cw, *cb, *g, *bb, *bm, *bv, *prelu, *qw, *qb, *kw, *kb, *vw,
        *vb, *uw, *ub, *gamma;
  } mp[2];
  for (int m = 0; m < 2; ++m) {
    int s = 2 + m * 16;
    mp[m].cw = (const float*)d_in[s + 0];
    mp[m].cb = (const float*)d_in[s + 1];
    mp[m].g = (const float*)d_in[s + 2];
    mp[m].bb = (const float*)d_in[s + 3];
    mp[m].bm = (const float*)d_in[s + 4];
    mp[m].bv = (const float*)d_in[s + 5];
    mp[m].prelu = (const float*)d_in[s + 6];
    mp[m].qw = (const float*)d_in[s + 7];
    mp[m].qb = (const float*)d_in[s + 8];
    mp[m].kw = (const float*)d_in[s + 9];
    mp[m].kb = (const float*)d_in[s + 10];
    mp[m].vw = (const float*)d_in[s + 11];
    mp[m].vb = (const float*)d_in[s + 12];
    mp[m].uw = (const float*)d_in[s + 13];
    mp[m].ub = (const float*)d_in[s + 14];
    mp[m].gamma = (const float*)d_in[s + 15];
  }

  // ---- carve workspace
  char* w = (char*)d_ws;
  size_t off = 0;
  auto alloc = [&](size_t bytes) -> void* {
    void* p = w + off;
    off = (off + bytes + 255) & ~(size_t)255;
    return p;
  };
  _Float16 *xt[2], *wconv[2], *wq[2], *wk[2], *wv[2], *wup[2];
  _Float16 *conv_t[2], *q_t[2], *k_t[2], *v_cm[2], *o_t[2];
  float *bnA[2], *bnB[2], *mst[2], *lst[2];
  _Float16* guard = (_Float16*)alloc((size_t)CIN * 2);
  for (int m = 0; m < 2; ++m) {
    xt[m] = (_Float16*)alloc((size_t)BATCH * NPIX * CIN * 2);
    wconv[m] = (_Float16*)alloc((size_t)9 * CD * CIN * 2);
    wq[m] = (_Float16*)alloc((size_t)CQ * CD * 2);
    wk[m] = (_Float16*)alloc((size_t)CQ * CD * 2);
    wv[m] = (_Float16*)alloc((size_t)CD * CD * 2);
    wup[m] = (_Float16*)alloc((size_t)CIN * CD * 2);
    conv_t[m] = (_Float16*)alloc((size_t)BATCH * NPIX * CD * 2);
    q_t[m] = (_Float16*)alloc((size_t)BATCH * NPIX * CQ * 2);
    k_t[m] = (_Float16*)alloc((size_t)BATCH * NPIX * CQ * 2);
    v_cm[m] = (_Float16*)alloc((size_t)BATCH * CD * NPIX * 2);
    o_t[m] = (_Float16*)alloc((size_t)BATCH * NPIX * CD * 2);
    bnA[m] = (float*)alloc((size_t)CD * 4);
    bnB[m] = (float*)alloc((size_t)CD * 4);
    mst[m] = (float*)alloc((size_t)BATCH * NPIX * 4);
    lst[m] = (float*)alloc((size_t)BATCH * NPIX * 4);
  }

  const int conv_blocks = BATCH * (CD / 16) * (NPIX / 16) * 32 / THREADS;
  const int gqk_blocks = BATCH * (CQ / 16) * (NPIX / 16) * 32 / THREADS;
  const int gv_blocks = BATCH * (CD / 16) * (NPIX / 16) * 32 / THREADS;
  const int st_blocks = BATCH * (NPIX / 16) * 32 / THREADS;
  const int up_blocks = BATCH * (CIN / 16) * (NPIX / 16) * 32 / THREADS;

  zero_guard_kernel<<<(CIN + THREADS - 1) / THREADS, THREADS, 0, stream>>>(
      guard, CIN);

  for (int m = 0; m < 2; ++m) {
    pack_x_kernel<<<(BATCH * CIN * NPIX) / THREADS, THREADS, 0, stream>>>(
        input[m], xt[m]);
    pack_wconv_kernel<<<(9 * CD * CIN) / THREADS, THREADS, 0, stream>>>(
        mp[m].cw, wconv[m]);
    pack_mat_kernel<<<(CQ * CD + THREADS - 1) / THREADS, THREADS, 0, stream>>>(
        mp[m].qw, wq[m], CQ * CD);
    pack_mat_kernel<<<(CQ * CD + THREADS - 1) / THREADS, THREADS, 0, stream>>>(
        mp[m].kw, wk[m], CQ * CD);
    pack_mat_kernel<<<(CD * CD + THREADS - 1) / THREADS, THREADS, 0, stream>>>(
        mp[m].vw, wv[m], CD * CD);
    pack_mat_kernel<<<(CIN * CD + THREADS - 1) / THREADS, THREADS, 0,
                      stream>>>(mp[m].uw, wup[m], CIN * CD);
    fuse_bn_kernel<<<1, THREADS, 0, stream>>>(mp[m].cb, mp[m].g, mp[m].bb,
                                              mp[m].bm, mp[m].bv, bnA[m],
                                              bnB[m]);
    conv_kernel<<<conv_blocks, THREADS, 0, stream>>>(
        xt[m], wconv[m], guard, bnA[m], bnB[m], mp[m].prelu, conv_t[m]);
    gemm_pm_kernel<<<gqk_blocks, THREADS, 0, stream>>>(wq[m], mp[m].qb,
                                                       conv_t[m], q_t[m], CQ,
                                                       CD);
    gemm_pm_kernel<<<gqk_blocks, THREADS, 0, stream>>>(wk[m], mp[m].kb,
                                                       conv_t[m], k_t[m], CQ,
                                                       CD);
    gemm_cm_kernel<<<gv_blocks, THREADS, 0, stream>>>(wv[m], mp[m].vb,
                                                      conv_t[m], v_cm[m], CD,
                                                      CD);
  }

  // attn_rgb: q_dsm (m=1) attends k_rgb (m=0); attn_dsm: q_rgb over k_dsm
  stats_kernel<<<st_blocks, THREADS, 0, stream>>>(q_t[1], k_t[0], mst[0],
                                                  lst[0]);
  stats_kernel<<<st_blocks, THREADS, 0, stream>>>(q_t[0], k_t[1], mst[1],
                                                  lst[1]);
  attn_o_kernel<<<BATCH * (NPIX / 16), THREADS, 0, stream>>>(
      q_t[1], k_t[0], v_cm[0], conv_t[0], mst[0], lst[0], mp[0].gamma, o_t[0]);
  attn_o_kernel<<<BATCH * (NPIX / 16), THREADS, 0, stream>>>(
      q_t[0], k_t[1], v_cm[1], conv_t[1], mst[1], lst[1], mp[1].gamma, o_t[1]);

  float* out = (float*)d_out;
  const size_t mod_stride = (size_t)BATCH * CIN * NPIX;
  up_kernel<<<up_blocks, THREADS, 0, stream>>>(wup[0], mp[0].ub, o_t[0],
                                               input[0], out);
  up_kernel<<<up_blocks, THREADS, 0, stream>>>(wup[1], mp[1].ub, o_t[1],
                                               input[1], out + mod_stride);
}
